// NodeCoordVAE_26594437496850
// MI455X (gfx1250) — matrix operand order repaired and checked by reading, not compiled, not run on hardware
//
#include <hip/hip_runtime.h>
#include <math.h>

#define HID   32
#define HEADS 4
#define LAT   16
#define EPS_BN 1e-5f

// fused qkv+skip layout: [q(128) | k(128) | v(256..) | r(384..415)], row stride 416
#define SQKV 416
#define OFF_Q 0
#define OFF_K 128
#define OFF_V 256
#define OFF_R 384
#define MAX_COLTILES 26   // fused layer: 416/16

typedef __attribute__((ext_vector_type(16))) _Float16 v16h;
typedef __attribute__((ext_vector_type(8)))  float    v8f;

// ---------------------------------------------------------------------------
// Pack W[cin, coutW] (cin<=32) into per-lane WMMA B fragments, f16, zero padded.
// frag[t]: t = tc*512 + lane*16 + j ; k = (lane>>4)*8 + (j&7) + (j>>3)*16 ; col = tc*16 + (lane&15)
// ---------------------------------------------------------------------------
__global__ void pack_b_kernel(const float* __restrict__ W, _Float16* __restrict__ frag,
                              int cin, int coutW, int total)
{
    int t = blockIdx.x * blockDim.x + threadIdx.x;
    if (t >= total) return;
    int tc   = t >> 9;
    int lane = (t >> 4) & 31;
    int j    = t & 15;
    int k    = (lane >> 4) * 8 + (j & 7) + (j >> 3) * 16;
    int col  = tc * 16 + (lane & 15);
    float v  = (k < cin && col < coutW) ? W[(size_t)k * coutW + col] : 0.0f;
    frag[t]  = (_Float16)v;
}

__global__ void pack_bias_kernel(float* __restrict__ dst, const float* __restrict__ src,
                                 int n_real, int total)
{
    int i = blockIdx.x * blockDim.x + threadIdx.x;
    if (i < total) dst[i] = (src && i < n_real) ? src[i] : 0.0f;
}

__global__ void pad_x_kernel(const float* __restrict__ x, float* __restrict__ xp, int n)
{
    int t = blockIdx.x * blockDim.x + threadIdx.x;
    if (t >= n * 32) return;
    int i = t >> 5, c = t & 31;
    xp[t] = (c < 3) ? x[i * 3 + c] : 0.0f;
}

// ---------------------------------------------------------------------------
// WMMA GEMM: Y[n, cout] = act(X[n, 32-padded] @ Wfrag + biasPad)
// Block stages the packed B fragments into LDS once (shared by its 8 waves);
// each wave owns a 16-row block: A loaded once (4x b128), per column tile
// one 32B LDS read (2x ds_load_b128) + one v_wmma_f32_16x16x32_f16.
// ---------------------------------------------------------------------------
__global__ void gemm_wmma_kernel(const float* __restrict__ X,
                                 const _Float16* __restrict__ frag,
                                 const float* __restrict__ bias,  // [colTiles*16] padded
                                 float* __restrict__ Y,
                                 int n, int cout, int colTiles, int relu)
{
    __shared__ _Float16 sfrag[MAX_COLTILES * 512];

    // cooperative staging: colTiles*1024 bytes, 16B chunks (done by ALL threads)
    int chunks = colTiles * 64;
    for (int i = threadIdx.x; i < chunks; i += blockDim.x)
        ((float4*)sfrag)[i] = ((const float4*)frag)[i];
    __syncthreads();

    int gid  = blockIdx.x * blockDim.x + threadIdx.x;
    int wave = gid >> 5;
    int lane = gid & 31;
    int rowTiles = (n + 15) >> 4;
    if (wave >= rowTiles) return;
    int hi  = lane >> 4;       // lane half
    int lid = lane & 15;
    int row = wave * 16 + lid;
    int rsafe = row < n ? row : n - 1;
    bool fullRows = (wave * 16 + 16) <= n;

    // A (16x32 f16): lanes 0-15 hold K={0..7,16..23}, lanes 16-31 K={8..15,24..31}
    const float* xr = X + (size_t)rsafe * 32;
    float4 a0 = *(const float4*)(xr + hi * 8);
    float4 a1 = *(const float4*)(xr + hi * 8 + 4);
    float4 a2 = *(const float4*)(xr + 16 + hi * 8);
    float4 a3 = *(const float4*)(xr + 16 + hi * 8 + 4);

    union { v16h v; _Float16 h[16]; } A;
    A.h[0]  = (_Float16)a0.x; A.h[1]  = (_Float16)a0.y;
    A.h[2]  = (_Float16)a0.z; A.h[3]  = (_Float16)a0.w;
    A.h[4]  = (_Float16)a1.x; A.h[5]  = (_Float16)a1.y;
    A.h[6]  = (_Float16)a1.z; A.h[7]  = (_Float16)a1.w;
    A.h[8]  = (_Float16)a2.x; A.h[9]  = (_Float16)a2.y;
    A.h[10] = (_Float16)a2.z; A.h[11] = (_Float16)a2.w;
    A.h[12] = (_Float16)a3.x; A.h[13] = (_Float16)a3.y;
    A.h[14] = (_Float16)a3.z; A.h[15] = (_Float16)a3.w;

    for (int tc = 0; tc < colTiles; ++tc) {
        v16h b = *(const v16h*)(sfrag + ((size_t)tc * 32 + lane) * 16);
        v8f c = {};
        c = __builtin_amdgcn_wmma_f32_16x16x32_f16(false, A.v, false, b,
                                                   (short)0, c, false, false);
        int col = tc * 16 + lid;
        if (col < cout) {
            float bi = bias[tc * 16 + lid];
            float* yp = Y + (size_t)(wave * 16 + hi * 8) * cout + col;
            if (fullRows) {
                #pragma unroll
                for (int r = 0; r < 8; ++r) {    // D: lanes0-15 M=r, lanes16-31 M=r+8
                    float val = c[r] + bi;
                    if (relu) val = fmaxf(val, 0.0f);
                    yp[(size_t)r * cout] = val;
                }
            } else {
                #pragma unroll
                for (int r = 0; r < 8; ++r) {
                    int orow = wave * 16 + hi * 8 + r;
                    if (orow < n) {
                        float val = c[r] + bi;
                        if (relu) val = fmaxf(val, 0.0f);
                        Y[(size_t)orow * cout + col] = val;
                    }
                }
            }
        }
    }
}

// ---------------------------------------------------------------------------
// Utility / graph kernels
// ---------------------------------------------------------------------------
__global__ void zero_kernel(float* p, int total)
{
    int i = blockIdx.x * blockDim.x + threadIdx.x;
    if (i < total) p[i] = 0.0f;
}

__device__ __forceinline__ unsigned f2ord(float f)
{
    unsigned u = __float_as_uint(f);
    return (u & 0x80000000u) ? ~u : (u | 0x80000000u);
}
__device__ __forceinline__ float ord2f(unsigned u)
{
    unsigned v = (u & 0x80000000u) ? (u ^ 0x80000000u) : ~u;
    return __uint_as_float(v);
}

__global__ void edge_logit_kernel(const float* __restrict__ qkv,
                                  const int* __restrict__ ei,
                                  float* __restrict__ logit, int E)
{
    int t = blockIdx.x * blockDim.x + threadIdx.x;
    if (t >= E * HEADS) return;
    int e = t >> 2, h = t & 3;
    int s = ei[e], d = ei[E + e];
    const float4* qp = (const float4*)(qkv + (size_t)d * SQKV + OFF_Q + h * HID);
    const float4* kp = (const float4*)(qkv + (size_t)s * SQKV + OFF_K + h * HID);
    float acc = 0.0f;
    #pragma unroll
    for (int i = 0; i < HID / 4; ++i) {
        float4 a = qp[i], b = kp[i];
        acc += a.x * b.x + a.y * b.y + a.z * b.z + a.w * b.w;
    }
    logit[t] = acc * 0.17677669529663687f;  // 1/sqrt(HID)
}

__global__ void segmax_init_kernel(unsigned* menc, int total)
{
    int i = blockIdx.x * blockDim.x + threadIdx.x;
    if (i < total) menc[i] = f2ord(-INFINITY);
}

__global__ void segmax_kernel(const float* __restrict__ logit,
                              const int* __restrict__ ei,
                              unsigned* __restrict__ menc, int E)
{
    int t = blockIdx.x * blockDim.x + threadIdx.x;
    if (t >= E * HEADS) return;
    int e = t >> 2, h = t & 3;
    atomicMax(&menc[ei[E + e] * HEADS + h], f2ord(logit[t]));
}

__global__ void segmax_fin_kernel(unsigned* menc, int total)
{
    int i = blockIdx.x * blockDim.x + threadIdx.x;
    if (i >= total) return;
    float m = ord2f(menc[i]);
    if (!(m >= -3.0e38f && m <= 3.0e38f)) m = 0.0f;  // isfinite guard
    ((float*)menc)[i] = m;
}

__global__ void edge_expsum_kernel(float* __restrict__ logit,
                                   const float* __restrict__ m,
                                   float* __restrict__ ssum,
                                   const int* __restrict__ ei, int E)
{
    int t = blockIdx.x * blockDim.x + threadIdx.x;
    if (t >= E * HEADS) return;
    int e = t >> 2, h = t & 3;
    int d = ei[E + e];
    float el = expf(logit[t] - m[d * HEADS + h]);
    logit[t] = el;
    atomicAdd(&ssum[d * HEADS + h], el);
}

__global__ void attn_agg_kernel(const float* __restrict__ expw,
                                const float* __restrict__ ssum,
                                const float* __restrict__ qkv,
                                const int* __restrict__ ei,
                                float* __restrict__ aout, int E)
{
    int t = blockIdx.x * blockDim.x + threadIdx.x;
    if (t >= E * HID) return;
    int e = t >> 5, c = t & (HID - 1);
    int s = ei[e], d = ei[E + e];
    float acc = 0.0f;
    #pragma unroll
    for (int h = 0; h < HEADS; ++h) {
        float alpha = expw[e * HEADS + h] / (ssum[d * HEADS + h] + 1e-16f);
        acc += alpha * qkv[(size_t)s * SQKV + OFF_V + h * HID + c];
    }
    atomicAdd(&aout[(size_t)d * HID + c], 0.25f * acc);  // head mean folded in
}

__global__ void tconv_combine_kernel(const float* __restrict__ aout,
                                     const float* __restrict__ qkv,  // r at OFF_R
                                     const float* __restrict__ betaW, // [96]
                                     float* __restrict__ out, int n)
{
    int i = blockIdx.x * blockDim.x + threadIdx.x;
    if (i >= n) return;
    const float* o  = aout + (size_t)i * HID;
    const float* rr = qkv + (size_t)i * SQKV + OFF_R;
    float acc = 0.0f;
    #pragma unroll
    for (int c = 0; c < HID; ++c) {
        float ov = o[c], rv = rr[c];
        acc += ov * betaW[c] + rv * betaW[HID + c] + (ov - rv) * betaW[2 * HID + c];
    }
    float beta = 1.0f / (1.0f + expf(-acc));
    #pragma unroll
    for (int c = 0; c < HID; ++c) {
        float val = beta * rr[c] + (1.0f - beta) * o[c];
        out[(size_t)i * HID + c] = fmaxf(val, 0.0f);   // relu(tconv(...))
    }
}

__global__ void bn_stats_kernel(const float* __restrict__ x,
                                float* __restrict__ stats, int n)
{
    __shared__ float ssum[256], ssq[256];
    int c = blockIdx.x;                       // one block per channel
    float s = 0.0f, q = 0.0f;
    for (int i = threadIdx.x; i < n; i += blockDim.x) {
        float v = x[(size_t)i * HID + c];
        s += v; q += v * v;
    }
    ssum[threadIdx.x] = s; ssq[threadIdx.x] = q;
    __syncthreads();
    for (int st = blockDim.x >> 1; st > 0; st >>= 1) {
        if (threadIdx.x < st) {
            ssum[threadIdx.x] += ssum[threadIdx.x + st];
            ssq[threadIdx.x]  += ssq[threadIdx.x + st];
        }
        __syncthreads();
    }
    if (threadIdx.x == 0) { stats[c] = ssum[0]; stats[HID + c] = ssq[0]; }
}

__global__ void bn_apply_kernel(float* __restrict__ x,
                                const float* __restrict__ stats,
                                const float* __restrict__ g,
                                const float* __restrict__ b, int n)
{
    int t = blockIdx.x * blockDim.x + threadIdx.x;
    if (t >= n * HID) return;
    int c = t & (HID - 1);
    float m   = stats[c] / (float)n;
    float var = stats[HID + c] / (float)n - m * m;
    x[t] = (x[t] - m) * rsqrtf(var + EPS_BN) * g[c] + b[c];
}

__global__ void deg_kernel(const int* __restrict__ ei, float* __restrict__ dinv, int E)
{
    int e = blockIdx.x * blockDim.x + threadIdx.x;
    if (e < E) atomicAdd(&dinv[ei[E + e]], 1.0f);
}

__global__ void dinv_kernel(float* dinv, int n)
{
    int i = blockIdx.x * blockDim.x + threadIdx.x;
    if (i < n) dinv[i] = rsqrtf(dinv[i] + 1.0f);   // (deg+1)^-0.5
}

__global__ void gcn_agg_kernel(const float* __restrict__ h,
                               const int* __restrict__ ei,
                               const float* __restrict__ dinv,
                               float* __restrict__ agg, int E, int cout)
{
    int t = blockIdx.x * blockDim.x + threadIdx.x;
    if (t >= E * cout) return;
    int e = t / cout, c = t - e * cout;
    int s = ei[e], d = ei[E + e];
    atomicAdd(&agg[(size_t)d * cout + c],
              dinv[s] * dinv[d] * h[(size_t)s * cout + c]);
}

__global__ void gcn_final_kernel(const float* __restrict__ h,
                                 const float* __restrict__ agg,
                                 const float* __restrict__ dinv,
                                 const float* __restrict__ bias,
                                 float* __restrict__ out, int n, int cout, int relu)
{
    int t = blockIdx.x * blockDim.x + threadIdx.x;
    if (t >= n * cout) return;
    int i = t / cout, c = t - i * cout;
    float di = dinv[i];
    float val = agg[t] + di * di * h[t] + bias[c];
    if (relu) val = fmaxf(val, 0.0f);
    out[t] = val;
}

// writes z into a zero-padded [n,32] buffer (upper 16 pre-zeroed)
__global__ void reparam_kernel(const float* __restrict__ mu,
                               const float* __restrict__ logvar,
                               const float* __restrict__ eps,
                               float* __restrict__ z, int n)
{
    int t = blockIdx.x * blockDim.x + threadIdx.x;
    if (t >= n * LAT) return;
    int i = t >> 4, c = t & (LAT - 1);
    z[(size_t)i * 32 + c] = mu[t] + eps[t] * expf(0.5f * logvar[t]);
}

__global__ void ln_kernel(const float* __restrict__ x,
                          const float* __restrict__ g,
                          const float* __restrict__ b,
                          float* __restrict__ y, int n)
{
    int i = blockIdx.x * blockDim.x + threadIdx.x;
    if (i >= n) return;
    const float* xr = x + (size_t)i * HID;
    float m = 0.0f;
    #pragma unroll
    for (int c = 0; c < HID; ++c) m += xr[c];
    m *= (1.0f / HID);
    float v = 0.0f;
    #pragma unroll
    for (int c = 0; c < HID; ++c) { float d = xr[c] - m; v += d * d; }
    v *= (1.0f / HID);
    float rs = rsqrtf(v + EPS_BN);
    #pragma unroll
    for (int c = 0; c < HID; ++c)
        y[(size_t)i * HID + c] = (xr[c] - m) * rs * g[c] + b[c];
}

// ---------------------------------------------------------------------------
// Host orchestration
// ---------------------------------------------------------------------------
static inline int blks(long long total, int bs) { return (int)((total + bs - 1) / bs); }

static void pack_w(const float* W, _Float16* frag, int cin, int coutW, int tileoff,
                   hipStream_t s)
{
    int total = ((coutW + 15) / 16) * 512;
    hipLaunchKernelGGL(pack_b_kernel, dim3(blks(total, 256)), dim3(256), 0, s,
                       W, frag + (size_t)tileoff * 512, cin, coutW, total);
}

static void pack_bias(float* dst, const float* src, int n_real, int total, hipStream_t s)
{
    hipLaunchKernelGGL(pack_bias_kernel, dim3(blks(total, 256)), dim3(256), 0, s,
                       dst, src, n_real, total);
}

static void gemm(const float* X, const _Float16* frag, const float* bias, float* Y,
                 int n, int cout, int relu, hipStream_t s)
{
    int colTiles = (cout + 15) / 16;
    long long threads = (long long)((n + 15) / 16) * 32;
    hipLaunchKernelGGL(gemm_wmma_kernel, dim3(blks(threads, 256)), dim3(256), 0, s,
                       X, frag, bias, Y, n, cout, colTiles, relu);
}

static void zero(float* p, long long total, hipStream_t s)
{
    hipLaunchKernelGGL(zero_kernel, dim3(blks(total, 256)), dim3(256), 0, s, p, (int)total);
}

extern "C" void kernel_launch(void* const* d_in, const int* in_sizes, int n_in,
                              void* d_out, int out_size, void* d_ws, size_t ws_size,
                              hipStream_t stream)
{
    const float* x   = (const float*)d_in[0];
    const int*   ei  = (const int*)d_in[1];
    const float* eps = (const float*)d_in[2];
    const int N = in_sizes[0] / 3;
    const int E = in_sizes[1] / 2;

    // params in setup_inputs() insertion order
    #define P(i) ((const float*)d_in[i])
    #define TC(l, j) P(3 + (l) * 9 + (j))   // 0 qW 1 qb 2 kW 3 kb 4 vW 5 vb 6 skW 7 skb 8 betaW
    const int BN0 = 3 + 4 * 9;              // 39..44: bn g,b x3
    const int GMU = BN0 + 6;                // 45,46
    const int GLV = GMU + 2;                // 47,48
    const int DG1 = GLV + 2;                // 49,50
    const int LN1 = DG1 + 2;                // 51,52
    const int DG2 = LN1 + 2;                // 53,54
    const int LN2 = DG2 + 2;                // 55,56
    const int ML1 = LN2 + 2;                // 57,58
    const int ML2 = ML1 + 2;                // 59,60

    // workspace layout (floats)
    float* f = (float*)d_ws;
    size_t off = 0;
    float* xpad  = f + off; off += (size_t)N * 32;
    float* qkv   = f + off; off += (size_t)N * SQKV;
    float* h     = f + off; off += (size_t)N * HID;
    float* aout  = f + off; off += (size_t)N * HID;
    float* tA    = f + off; off += (size_t)N * HID;
    float* tB    = f + off; off += (size_t)N * HID;
    float* z     = f + off; off += (size_t)N * 32;   // zero-padded latent
    float* logit = f + off; off += (size_t)E * HEADS;
    float* ssum  = f + off; off += (size_t)N * HEADS;
    unsigned* menc = (unsigned*)(f + off); off += (size_t)N * HEADS;
    float* dinv  = f + off; off += (size_t)N;
    float* stats = f + off; off += 2 * HID;
    _Float16* frag = (_Float16*)(f + off); off += MAX_COLTILES * 512 / 2 + 256;
    float* biasbuf = f + off; off += 512;

    float* out_recon = (float*)d_out;                       // [N,3]
    float* out_mu    = out_recon + (size_t)N * 3;           // [N,16]
    float* out_lv    = out_mu    + (size_t)N * LAT;         // [N,16]

    // degrees (shared by all GCN layers) + padded inputs
    zero(dinv, N, stream);
    hipLaunchKernelGGL(deg_kernel,  dim3(blks(E, 256)), dim3(256), 0, stream, ei, dinv, E);
    hipLaunchKernelGGL(dinv_kernel, dim3(blks(N, 256)), dim3(256), 0, stream, dinv, N);
    hipLaunchKernelGGL(pad_x_kernel, dim3(blks((long long)N * 32, 256)), dim3(256), 0,
                       stream, x, xpad, N);
    zero(z, (long long)N * 32, stream);   // upper 16 channels stay zero

    // ---- 4 TransformerConv layers (fused q|k|v|skip GEMM, cout=416) ----
    const float* cur = xpad;
    int cin = 3;
    for (int l = 0; l < 4; ++l) {
        pack_w(TC(l, 0), frag, cin, HEADS * HID, 0,  stream);   // q -> tiles 0..7
        pack_w(TC(l, 2), frag, cin, HEADS * HID, 8,  stream);   // k -> tiles 8..15
        pack_w(TC(l, 4), frag, cin, HEADS * HID, 16, stream);   // v -> tiles 16..23
        pack_w(TC(l, 6), frag, cin, HID,         24, stream);   // skip -> tiles 24..25
        pack_bias(biasbuf,       TC(l, 1), HEADS * HID, HEADS * HID, stream);
        pack_bias(biasbuf + 128, TC(l, 3), HEADS * HID, HEADS * HID, stream);
        pack_bias(biasbuf + 256, TC(l, 5), HEADS * HID, HEADS * HID, stream);
        pack_bias(biasbuf + 384, TC(l, 7), HID,          HID,         stream);
        gemm(cur, frag, biasbuf, qkv, N, SQKV, 0, stream);

        hipLaunchKernelGGL(edge_logit_kernel, dim3(blks((long long)E * HEADS, 256)),
                           dim3(256), 0, stream, qkv, ei, logit, E);
        hipLaunchKernelGGL(segmax_init_kernel, dim3(blks((long long)N * HEADS, 256)),
                           dim3(256), 0, stream, menc, N * HEADS);
        hipLaunchKernelGGL(segmax_kernel, dim3(blks((long long)E * HEADS, 256)),
                           dim3(256), 0, stream, logit, ei, menc, E);
        hipLaunchKernelGGL(segmax_fin_kernel, dim3(blks((long long)N * HEADS, 256)),
                           dim3(256), 0, stream, menc, N * HEADS);
        zero(ssum, (long long)N * HEADS, stream);
        hipLaunchKernelGGL(edge_expsum_kernel, dim3(blks((long long)E * HEADS, 256)),
                           dim3(256), 0, stream, logit, (const float*)menc, ssum, ei, E);
        zero(aout, (long long)N * HID, stream);
        hipLaunchKernelGGL(attn_agg_kernel, dim3(blks((long long)E * HID, 256)),
                           dim3(256), 0, stream, logit, ssum, qkv, ei, aout, E);
        hipLaunchKernelGGL(tconv_combine_kernel, dim3(blks(N, 256)),
                           dim3(256), 0, stream, aout, qkv, TC(l, 8), h, N);
        if (l < 3) {
            hipLaunchKernelGGL(bn_stats_kernel, dim3(HID), dim3(256), 0, stream, h, stats, N);
            hipLaunchKernelGGL(bn_apply_kernel, dim3(blks((long long)N * HID, 256)),
                               dim3(256), 0, stream, h, stats,
                               P(BN0 + 2 * l), P(BN0 + 2 * l + 1), N);
        }
        cur = h;
        cin = HID;
    }

    // ---- GCN mu / logvar ----
    pack_w(P(GMU), frag, HID, LAT, 0, stream);
    pack_bias(biasbuf, NULL, 0, LAT, stream);
    gemm(h, frag, biasbuf, tA, N, LAT, 0, stream);
    zero(tB, (long long)N * LAT, stream);
    hipLaunchKernelGGL(gcn_agg_kernel, dim3(blks((long long)E * LAT, 256)), dim3(256), 0,
                       stream, tA, ei, dinv, tB, E, LAT);
    hipLaunchKernelGGL(gcn_final_kernel, dim3(blks((long long)N * LAT, 256)), dim3(256), 0,
                       stream, tA, tB, dinv, P(GMU + 1), out_mu, N, LAT, 0);

    pack_w(P(GLV), frag, HID, LAT, 0, stream);
    gemm(h, frag, biasbuf, tA, N, LAT, 0, stream);
    zero(tB, (long long)N * LAT, stream);
    hipLaunchKernelGGL(gcn_agg_kernel, dim3(blks((long long)E * LAT, 256)), dim3(256), 0,
                       stream, tA, ei, dinv, tB, E, LAT);
    hipLaunchKernelGGL(gcn_final_kernel, dim3(blks((long long)N * LAT, 256)), dim3(256), 0,
                       stream, tA, tB, dinv, P(GLV + 1), out_lv, N, LAT, 0);

    hipLaunchKernelGGL(reparam_kernel, dim3(blks((long long)N * LAT, 256)), dim3(256), 0,
                       stream, out_mu, out_lv, eps, z, N);

    // ---- decoder GCN 1 (z: [N,32] zero-padded, cin=16) ----
    pack_w(P(DG1), frag, LAT, HID, 0, stream);
    pack_bias(biasbuf, NULL, 0, HID, stream);
    gemm(z, frag, biasbuf, tA, N, HID, 0, stream);
    zero(tB, (long long)N * HID, stream);
    hipLaunchKernelGGL(gcn_agg_kernel, dim3(blks((long long)E * HID, 256)), dim3(256), 0,
                       stream, tA, ei, dinv, tB, E, HID);
    hipLaunchKernelGGL(gcn_final_kernel, dim3(blks((long long)N * HID, 256)), dim3(256), 0,
                       stream, tA, tB, dinv, P(DG1 + 1), tB, N, HID, 1);
    hipLaunchKernelGGL(ln_kernel, dim3(blks(N, 256)), dim3(256), 0, stream,
                       tB, P(LN1), P(LN1 + 1), h, N);

    // ---- decoder GCN 2 ----
    pack_w(P(DG2), frag, HID, HID, 0, stream);
    gemm(h, frag, biasbuf, tA, N, HID, 0, stream);
    zero(tB, (long long)N * HID, stream);
    hipLaunchKernelGGL(gcn_agg_kernel, dim3(blks((long long)E * HID, 256)), dim3(256), 0,
                       stream, tA, ei, dinv, tB, E, HID);
    hipLaunchKernelGGL(gcn_final_kernel, dim3(blks((long long)N * HID, 256)), dim3(256), 0,
                       stream, tA, tB, dinv, P(DG2 + 1), tB, N, HID, 1);
    hipLaunchKernelGGL(ln_kernel, dim3(blks(N, 256)), dim3(256), 0, stream,
                       tB, P(LN2), P(LN2 + 1), h, N);

    // ---- MLP head ----
    pack_w(P(ML1), frag, HID, HID, 0, stream);
    pack_bias(biasbuf, P(ML1 + 1), HID, HID, stream);
    gemm(h, frag, biasbuf, tA, N, HID, 1, stream);

    pack_w(P(ML2), frag, HID, 3, 0, stream);
    pack_bias(biasbuf, P(ML2 + 1), 3, 16, stream);
    gemm(tA, frag, biasbuf, out_recon, N, 3, 0, stream);

    (void)n_in; (void)out_size; (void)ws_size; (void)off;
    #undef TC
    #undef P
}
